// ModelSquad_17420387353061
// MI455X (gfx1250) — matrix operand order repaired and checked by reading, not compiled, not run on hardware
//
#include <hip/hip_runtime.h>
#include <hip/hip_bf16.h>
#include <cstdint>
#include <cstddef>

// ---------------------------------------------------------------------------
// DMN episodic attention + AttentionGRU + span heads, for MI455X (gfx1250).
// B=64, T=512, H=256.
//
// All big GEMM families (folded W1 attention MLP, fact@Wr / fact@Wh
// precompute, per-step h@Ur / (r*h)@Uh inside the GRU scan) run on
// v_wmma_f32_16x16x32_bf16 (f32 accumulate).  The scan packs 16 batches per
// workgroup (M=16 tiles, full WMMA utilization), 4 workgroups, with bf16
// Ur^T/Uh^T staged into LDS (256KB of the 320KB WGP LDS) via the Tensor Data
// Mover (tensor_load_to_lds + s_wait_tensorcnt) when the toolchain exposes
// the builtin.
// ---------------------------------------------------------------------------

typedef __attribute__((ext_vector_type(16))) __bf16 v16bf;
typedef __attribute__((ext_vector_type(8)))  float  v8f;

#define Hd 256
#define Td 512
#define Bd 64

#ifdef __has_builtin
#if __has_builtin(__builtin_amdgcn_tensor_load_to_lds) && \
    __has_builtin(__builtin_amdgcn_s_wait_tensorcnt)
#define HAVE_TDM 1
#endif
#endif

static __device__ __forceinline__ __bf16 bf(float x) { return (__bf16)x; }

static __device__ __forceinline__ v8f wmma_bf16(v16bf a, v16bf b, v8f c) {
  // D = A(16x32 bf16) * B(32x16 bf16) + C(16x16 f32)
  return __builtin_amdgcn_wmma_f32_16x16x32_bf16(false, a, false, b,
                                                 (short)0, c, false, false);
}

// Build a 16x32 bf16 A-fragment from an f32 row-major 16x256 block.
// Contiguous K order (matches B): lanes 0-15 (M=lane) hold K = k0..k0+15
// with k0 = kt*32; lanes 16-31 (M=lane-16) hold K = k0+16..k0+31.
// => per lane: 16 consecutive K at kt*32 + hi*16. All 16B-aligned.
static __device__ __forceinline__ void buildA_f(const float* __restrict__ fb,
                                                int hi, v16bf A[8]) {
#pragma unroll
  for (int kt = 0; kt < 8; ++kt) {
    const int k0 = kt * 32 + hi * 16;
    const float4 f0 = *(const float4*)(fb + k0);
    const float4 f1 = *(const float4*)(fb + k0 + 4);
    const float4 f2 = *(const float4*)(fb + k0 + 8);
    const float4 f3 = *(const float4*)(fb + k0 + 12);
    const float fv[16] = {f0.x, f0.y, f0.z, f0.w, f1.x, f1.y, f1.z, f1.w,
                          f2.x, f2.y, f2.z, f2.w, f3.x, f3.y, f3.z, f3.w};
#pragma unroll
    for (int i = 0; i < 16; ++i) A[kt][i] = bf(fv[i]);
  }
}

// ---------------------------------------------------------------------------
// Prep: fold W1 (m==q => 3 effective HxH blocks), transpose all GEMM weights
// to bf16 N-major (so WMMA B fragments are contiguous 16-bf16 runs per lane).
// ---------------------------------------------------------------------------
__global__ __launch_bounds__(256) void k_prep_weights(
    const float* __restrict__ W1, const float* __restrict__ Wr,
    const float* __restrict__ Wh, const float* __restrict__ Ur,
    const float* __restrict__ Uh, __bf16* __restrict__ WT1,
    __bf16* __restrict__ WrT, __bf16* __restrict__ WhT,
    __bf16* __restrict__ UrT, __bf16* __restrict__ UhT) {
  const int n = blockIdx.x, k = threadIdx.x;
  WT1[0 * Hd * Hd + n * Hd + k] = bf(W1[(0 * Hd + k) * Hd + n]);
  WT1[1 * Hd * Hd + n * Hd + k] =
      bf(W1[(3 * Hd + k) * Hd + n] + W1[(4 * Hd + k) * Hd + n]);
  WT1[2 * Hd * Hd + n * Hd + k] =
      bf(W1[(5 * Hd + k) * Hd + n] + W1[(6 * Hd + k) * Hd + n]);
  WrT[n * Hd + k] = bf(Wr[k * Hd + n]);
  WhT[n * Hd + k] = bf(Wh[k * Hd + n]);
  UrT[n * Hd + k] = bf(Ur[k * Hd + n]);
  UhT[n * Hd + k] = bf(Uh[k * Hd + n]);
}

// qbias[b][n] = b1[n] + q_b @ (W1_m + W1_q)  (tiny, f32)
__global__ __launch_bounds__(256) void k_prep_qbias(
    const float* __restrict__ q, const float* __restrict__ W1,
    const float* __restrict__ b1, float* __restrict__ qbias) {
  const int b = blockIdx.x, n = threadIdx.x;
  float s = b1[n];
  for (int k = 0; k < Hd; ++k)
    s += q[b * Hd + k] * (W1[(Hd + k) * Hd + n] + W1[(2 * Hd + k) * Hd + n]);
  qbias[b * Hd + n] = s;
}

// qs[b] = bs + q_b . Ws[0:H];  qe[b] = be + q_b . We[0:H]
__global__ void k_prep_qse(const float* __restrict__ q,
                           const float* __restrict__ Ws,
                           const float* __restrict__ bs,
                           const float* __restrict__ We,
                           const float* __restrict__ be,
                           float* __restrict__ qs, float* __restrict__ qe) {
  const int b = threadIdx.x;
  float s = bs[0], e = be[0];
  for (int k = 0; k < Hd; ++k) {
    const float qv = q[b * Hd + k];
    s += qv * Ws[k];
    e += qv * We[k];
  }
  qs[b] = s;
  qe[b] = e;
}

// ---------------------------------------------------------------------------
// Attention logits: att[b,t] = b2 + sum_n W2[n]*tanh(hidden[b,t,n]),
// hidden = f@W1f + (f*q)@W1fq + |f-q|@W1fa + qbias[b].
// One wave per M-tile of 16 (b,t) rows; 3 mats x 8 ktiles WMMA per ntile,
// fused tanh+W2 reduction in the D-fragment layout (no 32MB hidden tensor).
// ---------------------------------------------------------------------------
__global__ __launch_bounds__(256) void k_attention(
    const float* __restrict__ fact, const float* __restrict__ q,
    const __bf16* __restrict__ WT1, const float* __restrict__ qbias,
    const float* __restrict__ W2, const float* __restrict__ b2,
    float* __restrict__ att) {
  const int tid = threadIdx.x;
  const int lane = tid & 31;
  const int ln = lane & 15;
  const int hi = lane >> 4;
  const int tile = blockIdx.x * 8 + (tid >> 5);  // 2048 M-tiles
  const int b = tile >> 5;                       // 32 tiles per batch (T/16)
  const int t0 = (tile & 31) * 16;

  const float* fb = fact + ((size_t)b * Td + t0 + ln) * Hd;  // A row = lane&15
  const float* qp = q + (size_t)b * Hd;

  v16bf A[3][8];
#pragma unroll
  for (int kt = 0; kt < 8; ++kt) {
    const int k0 = kt * 32 + hi * 16;
    const float4 f0 = *(const float4*)(fb + k0);
    const float4 f1 = *(const float4*)(fb + k0 + 4);
    const float4 f2 = *(const float4*)(fb + k0 + 8);
    const float4 f3 = *(const float4*)(fb + k0 + 12);
    const float4 q0 = *(const float4*)(qp + k0);
    const float4 q1 = *(const float4*)(qp + k0 + 4);
    const float4 q2 = *(const float4*)(qp + k0 + 8);
    const float4 q3 = *(const float4*)(qp + k0 + 12);
    const float fv[16] = {f0.x, f0.y, f0.z, f0.w, f1.x, f1.y, f1.z, f1.w,
                          f2.x, f2.y, f2.z, f2.w, f3.x, f3.y, f3.z, f3.w};
    const float qv[16] = {q0.x, q0.y, q0.z, q0.w, q1.x, q1.y, q1.z, q1.w,
                          q2.x, q2.y, q2.z, q2.w, q3.x, q3.y, q3.z, q3.w};
#pragma unroll
    for (int i = 0; i < 16; ++i) {
      A[0][kt][i] = bf(fv[i]);
      A[1][kt][i] = bf(fv[i] * qv[i]);
      A[2][kt][i] = bf(fabsf(fv[i] - qv[i]));
    }
  }

  float part[8] = {0.f, 0.f, 0.f, 0.f, 0.f, 0.f, 0.f, 0.f};
  for (int nt = 0; nt < 16; ++nt) {
    v8f acc = {};
#pragma unroll
    for (int mat = 0; mat < 3; ++mat) {
      const __bf16* wp = WT1 + mat * Hd * Hd + ((nt * 16 + ln) * Hd + hi * 16);
#pragma unroll
      for (int kt = 0; kt < 8; ++kt) {
        const v16bf bm = *(const v16bf*)(wp + kt * 32);
        acc = wmma_bf16(A[mat][kt], bm, acc);
      }
    }
    const int n = nt * 16 + ln;
    const float w2n = W2[n];
    const float qbn = qbias[b * Hd + n];
#pragma unroll
    for (int v = 0; v < 8; ++v) part[v] += tanhf(acc[v] + qbn) * w2n;
  }
  // reduce over the 16 N-lanes of each half (masks < 16 stay within a half)
#pragma unroll
  for (int v = 0; v < 8; ++v) {
    part[v] += __shfl_xor(part[v], 1, 32);
    part[v] += __shfl_xor(part[v], 2, 32);
    part[v] += __shfl_xor(part[v], 4, 32);
    part[v] += __shfl_xor(part[v], 8, 32);
  }
  if (ln == 0) {
    const float b2v = b2[0];
#pragma unroll
    for (int v = 0; v < 8; ++v)
      att[b * Td + t0 + v + 8 * hi] = part[v] + b2v;
  }
}

// softmax over T per batch; store transposed gT[t][b] for scan locality
__global__ __launch_bounds__(512) void k_softmax(const float* __restrict__ att,
                                                 float* __restrict__ gT) {
  __shared__ float red[512];
  const int b = blockIdx.x, t = threadIdx.x;
  const float v = att[b * Td + t];
  red[t] = v;
  __syncthreads();
  for (int s = 256; s > 0; s >>= 1) {
    if (t < s) red[t] = fmaxf(red[t], red[t + s]);
    __syncthreads();
  }
  const float mx = red[0];
  __syncthreads();
  const float e = __expf(v - mx);
  red[t] = e;
  __syncthreads();
  for (int s = 256; s > 0; s >>= 1) {
    if (t < s) red[t] += red[t + s];
    __syncthreads();
  }
  gT[t * Bd + b] = e / red[0];
}

// ---------------------------------------------------------------------------
// XR = fact@Wr + br, XH = fact@Wh + bh, stored bf16 [T][B][H] so each scan
// step touches one contiguous 8KB-per-WG slab.
// ---------------------------------------------------------------------------
static __device__ __forceinline__ void xw_store(const v16bf A[8],
                                                const __bf16* __restrict__ WT,
                                                const float* __restrict__ bias,
                                                __bf16* __restrict__ out,
                                                int ln, int hi, int b, int t0) {
  for (int nt = 0; nt < 16; ++nt) {
    v8f acc = {};
    const __bf16* wp = WT + ((nt * 16 + ln) * Hd + hi * 16);
#pragma unroll
    for (int kt = 0; kt < 8; ++kt) {
      const v16bf bm = *(const v16bf*)(wp + kt * 32);
      acc = wmma_bf16(A[kt], bm, acc);
    }
    const int n = nt * 16 + ln;
    const float bi = bias[n];
#pragma unroll
    for (int v = 0; v < 8; ++v) {
      const int t = t0 + v + 8 * hi;  // D layout: row = v + 8*(lane>>4)
      out[((size_t)t * Bd + b) * Hd + n] = bf(acc[v] + bi);
    }
  }
}

__global__ __launch_bounds__(256) void k_xrh(
    const float* __restrict__ fact, const __bf16* __restrict__ WrT,
    const __bf16* __restrict__ WhT, const float* __restrict__ br,
    const float* __restrict__ bh, __bf16* __restrict__ XR,
    __bf16* __restrict__ XH) {
  const int tid = threadIdx.x;
  const int lane = tid & 31;
  const int ln = lane & 15;
  const int hi = lane >> 4;
  const int tile = blockIdx.x * 8 + (tid >> 5);
  const int b = tile >> 5;
  const int t0 = (tile & 31) * 16;

  const float* fb = fact + ((size_t)b * Td + t0 + ln) * Hd;
  v16bf A[8];
  buildA_f(fb, hi, A);
  xw_store(A, WrT, br, XR, ln, hi, b, t0);
  xw_store(A, WhT, bh, XH, ln, hi, b, t0);
}

// ---------------------------------------------------------------------------
// AttentionGRU scan. 4 workgroups x 512 threads (16 waves); WG w owns batches
// 16w..16w+15 (=> M=16 WMMA tiles, full utilization). Wave i owns N-tile i.
// LDS: Ur^T + Uh^T bf16 (256KB) + h + r*h f32 tiles (32KB) = 288KB dynamic.
// Weights staged via TDM (tensor_load_to_lds, TENSORcnt) when available.
// Per step: r = sigmoid(h@Ur^T + XR); hc = tanh((r*h)@Uh^T + XH);
// h = valid ? g*hc+(1-g)*h : h; E = valid ? h_new : 0.
// ---------------------------------------------------------------------------
__global__ __launch_bounds__(512) void k_scan(
    const __bf16* __restrict__ UrUhT,  // UrT immediately followed by UhT
    const __bf16* __restrict__ XR, const __bf16* __restrict__ XH,
    const float* __restrict__ gT, const int* __restrict__ input_len,
    __bf16* __restrict__ E) {
  extern __shared__ char smem[];
  __bf16* sU = (__bf16*)smem;                                   // 2*H*H bf16
  float* hbuf = (float*)(smem + 2 * Hd * Hd * sizeof(__bf16));  // 16*H f32
  float* rhbuf = hbuf + 16 * Hd;                                // 16*H f32

  const int tid = threadIdx.x;
  const int lane = tid & 31;
  const int ln = lane & 15;
  const int hi = lane >> 4;
  const int wave = tid >> 5;  // 0..15 => N-tile
  const int n = wave * 16 + ln;
  const int b0 = blockIdx.x * 16;

#if defined(HAVE_TDM)
  // Stage Ur^T+Uh^T (256KB) with one Tensor Data Mover descriptor, issued by
  // wave 0 only.  D# describes a 1-row tensor of 32768 8-byte elements.
  if (tid < 32) {
    typedef __attribute__((ext_vector_type(4))) unsigned int u32x4;
    typedef __attribute__((ext_vector_type(4))) int i32x4;
    typedef __attribute__((ext_vector_type(8))) int i32x8;
    const unsigned long long ga = (unsigned long long)(size_t)UrUhT;
    const unsigned lds_base = __builtin_amdgcn_groupstaticsize();  // dyn base
    u32x4 g0;
    g0[0] = 1u;                                  // count=1 (valid), user mode
    g0[1] = lds_base;                            // lds_addr (bytes)
    g0[2] = (unsigned)(ga & 0xFFFFFFFFu);        // global_addr[31:0]
    g0[3] = (unsigned)((ga >> 32) & 0x01FFFFFFu) | (2u << 30);  // addr, type=2
    i32x8 g1;
    g1[0] = (3 << 16);            // workgroup_mask=0, data_size=3 (8B)
    g1[1] = (int)(32768u << 16);  // tensor_dim0[15:0]=32768 (bits 63:48)
    g1[2] = (int)(1u << 16);      // tensor_dim0[31:16]=0; tensor_dim1 lo=1
    g1[3] = (int)(32768u << 16);  // tensor_dim1 hi=0; tile_dim0=32768
    g1[4] = 1;                    // tile_dim1=1, tile_dim2=0
    g1[5] = 32768;                // tensor_dim0_stride lo
    g1[6] = 0;                    // stride0 hi=0, tensor_dim1_stride lo16=0
    g1[7] = 0;
    const i32x4 gz = {0, 0, 0, 0};
#if __clang_major__ >= 23
    const i32x8 gz8 = {0, 0, 0, 0, 0, 0, 0, 0};
    __builtin_amdgcn_tensor_load_to_lds(g0, g1, gz, gz, gz8, 0);
#else
    __builtin_amdgcn_tensor_load_to_lds(g0, g1, gz, gz, 0);
#endif
    __builtin_amdgcn_s_wait_tensorcnt(0);  // s_wait_tensorcnt 0
  }
#else
  {  // fallback: stage Ur^T + Uh^T into LDS with b128 copies
    const uint4* src = (const uint4*)UrUhT;
    uint4* dst = (uint4*)sU;
    for (int i = tid; i < (2 * Hd * Hd * 2) / 16; i += 512) dst[i] = src[i];
  }
#endif
  for (int i = tid; i < 16 * Hd; i += 512) {
    hbuf[i] = 0.f;
    rhbuf[i] = 0.f;
  }
  int lenv[8];
#pragma unroll
  for (int v = 0; v < 8; ++v) lenv[v] = input_len[b0 + v + 8 * hi];
  __syncthreads();

  const __bf16* sUr = sU;
  const __bf16* sUh = sU + Hd * Hd;

  for (int t = 0; t < Td; ++t) {
    // ---- phase 1: r = sigmoid(h @ Ur^T + XR) ; rh = r*h ----
    v8f acc = {};
#pragma unroll
    for (int kt = 0; kt < 8; ++kt) {
      const int k0 = kt * 32 + hi * 16;
      v16bf a;
#pragma unroll
      for (int i = 0; i < 8; ++i) {
        a[i] = bf(hbuf[ln * Hd + k0 + i]);
        a[8 + i] = bf(hbuf[ln * Hd + k0 + 8 + i]);
      }
      const v16bf bm = *(const v16bf*)(sUr + n * Hd + kt * 32 + hi * 16);
      acc = wmma_bf16(a, bm, acc);
    }
    float hval[8];
#pragma unroll
    for (int v = 0; v < 8; ++v) {
      const int m = v + 8 * hi;
      const float xr = (float)XR[((size_t)t * Bd + b0 + m) * Hd + n];
      hval[v] = hbuf[m * Hd + n];
      const float r = 1.f / (1.f + __expf(-(acc[v] + xr)));
      rhbuf[m * Hd + n] = r * hval[v];
    }
    __syncthreads();

    // ---- phase 2: hc = tanh(rh @ Uh^T + XH); blend with gate ----
    v8f acc2 = {};
#pragma unroll
    for (int kt = 0; kt < 8; ++kt) {
      const int k0 = kt * 32 + hi * 16;
      v16bf a;
#pragma unroll
      for (int i = 0; i < 8; ++i) {
        a[i] = bf(rhbuf[ln * Hd + k0 + i]);
        a[8 + i] = bf(rhbuf[ln * Hd + k0 + 8 + i]);
      }
      const v16bf bm = *(const v16bf*)(sUh + n * Hd + kt * 32 + hi * 16);
      acc2 = wmma_bf16(a, bm, acc2);
    }
    if (t + 1 < Td) {  // pull next timestep slab (global_prefetch_b8)
      __builtin_prefetch(&XR[((size_t)(t + 1) * Bd + b0) * Hd], 0, 1);
      __builtin_prefetch(&XH[((size_t)(t + 1) * Bd + b0) * Hd], 0, 1);
    }
#pragma unroll
    for (int v = 0; v < 8; ++v) {
      const int m = v + 8 * hi;
      const size_t idx = ((size_t)t * Bd + b0 + m) * Hd + n;
      const float xh = (float)XH[idx];
      const float hc = tanhf(acc2[v] + xh);
      const float g = gT[(size_t)t * Bd + b0 + m];
      const float hn = g * hc + (1.f - g) * hval[v];
      const bool valid = (t < lenv[v]);
      hbuf[m * Hd + n] = valid ? hn : hval[v];
      E[idx] = bf(valid ? hn : 0.f);
    }
    __syncthreads();
  }
}

// ---------------------------------------------------------------------------
// out_s[b,t] = tanh(qs[b] + E[b,t,:].Ws[H:2H]); out_e analogous.
// One wave per (b,t) row; coalesced bf16 loads; full-wave xor reduction.
// ---------------------------------------------------------------------------
__global__ __launch_bounds__(256) void k_final(
    const __bf16* __restrict__ E, const float* __restrict__ Ws,
    const float* __restrict__ We, const float* __restrict__ qs,
    const float* __restrict__ qe, float* __restrict__ out) {
  const int tid = threadIdx.x, lane = tid & 31;
  const int row = blockIdx.x * 8 + (tid >> 5);
  const int b = row >> 9;   // / T
  const int t = row & 511;  // % T
  const __bf16* ep = E + ((size_t)t * Bd + b) * Hd;
  float ps = 0.f, pe = 0.f;
#pragma unroll
  for (int i = 0; i < 8; ++i) {
    const int n = lane * 8 + i;
    const float ev = (float)ep[n];
    ps += ev * Ws[Hd + n];
    pe += ev * We[Hd + n];
  }
#pragma unroll
  for (int m = 1; m < 32; m <<= 1) {
    ps += __shfl_xor(ps, m, 32);
    pe += __shfl_xor(pe, m, 32);
  }
  if (lane == 0) {
    out[(size_t)b * Td + t] = tanhf(qs[b] + ps);
    out[(size_t)Bd * Td + (size_t)b * Td + t] = tanhf(qe[b] + pe);
  }
}

// ---------------------------------------------------------------------------
extern "C" void kernel_launch(void* const* d_in, const int* in_sizes, int n_in,
                              void* d_out, int out_size, void* d_ws,
                              size_t ws_size, hipStream_t stream) {
  (void)in_sizes; (void)n_in; (void)out_size; (void)ws_size;
  const float* q    = (const float*)d_in[0];
  const float* fact = (const float*)d_in[1];
  const int*   len  = (const int*)d_in[2];
  const float* W1   = (const float*)d_in[3];
  const float* b1   = (const float*)d_in[4];
  const float* W2   = (const float*)d_in[5];
  const float* b2   = (const float*)d_in[6];
  const float* Wr   = (const float*)d_in[7];
  const float* Ur   = (const float*)d_in[8];
  const float* br   = (const float*)d_in[9];
  const float* Wh   = (const float*)d_in[10];
  const float* Uh   = (const float*)d_in[11];
  const float* bh   = (const float*)d_in[12];
  const float* Ws   = (const float*)d_in[13];
  const float* bs   = (const float*)d_in[14];
  const float* We   = (const float*)d_in[15];
  const float* be   = (const float*)d_in[16];

  char* ws = (char*)d_ws;  // all offsets 256B-aligned; total ~49.2 MB
  __bf16* WT1   = (__bf16*)(ws + 0);         // 3*H*H bf16   = 393216 B
  __bf16* WrT   = (__bf16*)(ws + 393216);    // H*H bf16     = 131072 B
  __bf16* WhT   = (__bf16*)(ws + 524288);
  __bf16* UrT   = (__bf16*)(ws + 655360);    // UhT must follow contiguously
  __bf16* UhT   = (__bf16*)(ws + 786432);
  float*  qbias = (float*)(ws + 917504);     // B*H f32      = 65536 B
  float*  qs    = (float*)(ws + 983040);     // 256 B
  float*  qe    = (float*)(ws + 983296);     // 256 B
  float*  att   = (float*)(ws + 983552);     // B*T f32      = 131072 B
  float*  gT    = (float*)(ws + 1114624);    // T*B f32      = 131072 B
  __bf16* XR    = (__bf16*)(ws + 1245696);   // T*B*H bf16   = 16777216 B
  __bf16* XH    = (__bf16*)(ws + 18022912);
  __bf16* E     = (__bf16*)(ws + 34800128);

  k_prep_weights<<<Hd, Hd, 0, stream>>>(W1, Wr, Wh, Ur, Uh, WT1, WrT, WhT,
                                        UrT, UhT);
  k_prep_qbias<<<Bd, Hd, 0, stream>>>(q, W1, b1, qbias);
  k_prep_qse<<<1, Bd, 0, stream>>>(q, Ws, bs, We, be, qs, qe);
  k_attention<<<256, 256, 0, stream>>>(fact, q, WT1, qbias, W2, b2, att);
  k_softmax<<<Bd, Td, 0, stream>>>(att, gT);
  k_xrh<<<256, 256, 0, stream>>>(fact, WrT, WhT, br, bh, XR, XH);
  k_scan<<<4, 512, 294912, stream>>>(UrT, XR, XH, gT, len, E);
  k_final<<<4096, 256, 0, stream>>>(E, Ws, We, qs, qe, (float*)d_out);
  (void)UhT;
}
